// LSTMCell_14027363189407
// MI455X (gfx1250) — matrix operand order repaired and checked by reading, not compiled, not run on hardware
//
#include <hip/hip_runtime.h>

typedef __attribute__((ext_vector_type(16))) __bf16 v16bf;
typedef __attribute__((ext_vector_type(8)))  __bf16 v8bf;
typedef __attribute__((ext_vector_type(4)))  __bf16 v4bf;
typedef __attribute__((ext_vector_type(8)))  float  v8f;
typedef __attribute__((ext_vector_type(4)))  float  v4f;
typedef __attribute__((ext_vector_type(4)))  int    v4i;

#define B_DIM 4096
#define DIN   1024
#define DH    1024
#define K_DIM 2048            // DIN + DH
#define N_DIM 4096            // 4 * DH (gate-interleaved: n = j*4 + g)
#define HN    (B_DIM * DH)    // elements per output tensor

// Async global->LDS path (CDNA5): guarded so either toolchain compiles.
#if defined(__gfx1250__) && __has_builtin(__builtin_amdgcn_global_load_async_to_lds_b128)
#define HAVE_ASYNC_LDS 1
#else
#define HAVE_ASYNC_LDS 0
#endif

#if defined(__gfx1250__) && __has_builtin(__builtin_amdgcn_s_wait_asynccnt)
#define WAIT_ASYNC(n) __builtin_amdgcn_s_wait_asynccnt(n)
#elif defined(__gfx1250__)
#define WAIT_ASYNC(n) asm volatile("s_wait_asynccnt " #n)
#else
#define WAIT_ASYNC(n)
#endif

// Builtin prototype (from hipcc diagnostic): src = int4 __device__* (AS1),
// dst = int4 __shared__* (AS3), then imm offset, imm cpol.
#define GPTR(p) ((__attribute__((address_space(1))) v4i*)(p))
#define LPTR(p) ((__attribute__((address_space(3))) v4i*)(p))

// ---------------------------------------------------------------------------
// Pack x||h (fp32) into a bf16 [B, 2048] row-major matrix.
// ---------------------------------------------------------------------------
__global__ __launch_bounds__(256) void pack_xh(const float* __restrict__ x,
                                               const float* __restrict__ h,
                                               __bf16* __restrict__ xh) {
    int idx = (blockIdx.x * blockDim.x + threadIdx.x) * 4;   // element in [0, B*K)
    int m = idx >> 11;          // / 2048
    int k = idx & 2047;
    const float* src = (k < DIN) ? (x + m * DIN + k) : (h + m * DH + (k - DIN));
    v4f v = *(const v4f*)src;
    v4bf o;
    o.x = (__bf16)v.x; o.y = (__bf16)v.y; o.z = (__bf16)v.z; o.w = (__bf16)v.w;
    *(v4bf*)(xh + idx) = o;
}

// ---------------------------------------------------------------------------
// Pack weights gate-INTERLEAVED into bf16 [2048, 4096]:
//   Wp[k][j*4 + g] = W_g[k][j]
// so every 128-wide N tile holds all four gates of 32 hidden columns.
// ---------------------------------------------------------------------------
__global__ __launch_bounds__(256) void pack_w(const float* __restrict__ Wf,
                                              const float* __restrict__ Wi,
                                              const float* __restrict__ Wc,
                                              const float* __restrict__ Wo,
                                              __bf16* __restrict__ Wp) {
    int idx = (blockIdx.x * blockDim.x + threadIdx.x) * 4;   // one j-quad
    int k = idx >> 12;          // / 4096
    int j = (idx & 4095) >> 2;
    int s = k * DH + j;
    v4bf o;
    o.x = (__bf16)Wf[s];
    o.y = (__bf16)Wi[s];
    o.z = (__bf16)Wc[s];
    o.w = (__bf16)Wo[s];
    *(v4bf*)(Wp + idx) = o;
}

// ---------------------------------------------------------------------------
// Fused bf16 WMMA GEMM + LSTM epilogue.
// Block tile 128x128 (M x interleaved-N), BK=32, double-buffered LDS fed by
// async global->LDS DMA. 8 waves in 2x4 grid; each wave: 4x2 wmma fragments.
// After the K loop the fp32 accumulator tile is staged in LDS (overlapping
// the A/B buffers) and consumed as per-column gate quads (f,i,c,o).
// ---------------------------------------------------------------------------
#define BM 128
#define BN 128
#define BK 32
#define LDA 40      // padded A row (elements)
#define LDB 136     // padded B row (elements)
#define NIT (K_DIM / BK)

#define AS_OFF(b) ((b) ? 10240 : 0)              // 2 x 10240 B
#define BS_OFF(b) ((b) ? 29184 : 20480)          // 2 x 8704 B
#define SMEM_BYTES 65536                         // Ct tile (128x128 fp32) reuses all of it

__device__ __forceinline__ float sigmoid_(float x) { return 1.f / (1.f + __expf(-x)); }
__device__ __forceinline__ float tanh_(float x) {
    float e = __expf(2.f * x);
    return 1.f - 2.f / (e + 1.f);
}

__device__ __forceinline__ void stage_tile(const __bf16* __restrict__ A,
                                           const __bf16* __restrict__ Bm,
                                           char* smem, int buf, int kt,
                                           int m0, int n0, int tid) {
    const int arow = tid >> 1;            // 0..127
    const int acol = (tid & 1) * 16;      // 0 / 16 elements
    const int brow = tid >> 3;            // 0..31
    const int bcol = (tid & 7) * 16;      // 0..112 elements

    const __bf16* ag = A  + (size_t)(m0 + arow) * K_DIM + kt + acol;
    const __bf16* bg = Bm + (size_t)(kt + brow) * N_DIM + n0 + bcol;
    __bf16* al = (__bf16*)(smem + AS_OFF(buf)) + arow * LDA + acol;
    __bf16* bl = (__bf16*)(smem + BS_OFF(buf)) + brow * LDB + bcol;

#if HAVE_ASYNC_LDS
    __builtin_amdgcn_global_load_async_to_lds_b128(GPTR(ag),     LPTR(al),     0, 0);
    __builtin_amdgcn_global_load_async_to_lds_b128(GPTR(ag + 8), LPTR(al + 8), 0, 0);
    __builtin_amdgcn_global_load_async_to_lds_b128(GPTR(bg),     LPTR(bl),     0, 0);
    __builtin_amdgcn_global_load_async_to_lds_b128(GPTR(bg + 8), LPTR(bl + 8), 0, 0);
#else
    v8bf a0 = *(const v8bf*)(ag);
    v8bf a1 = *(const v8bf*)(ag + 8);
    v8bf b0 = *(const v8bf*)(bg);
    v8bf b1 = *(const v8bf*)(bg + 8);
    *(v8bf*)(al)     = a0;
    *(v8bf*)(al + 8) = a1;
    *(v8bf*)(bl)     = b0;
    *(v8bf*)(bl + 8) = b1;
#endif
}

__global__ __launch_bounds__(256) void gemm_lstm(const __bf16* __restrict__ A,
                                                 const __bf16* __restrict__ Bm,
                                                 const float* __restrict__ c_old,
                                                 const float* __restrict__ b_f,
                                                 const float* __restrict__ b_i,
                                                 const float* __restrict__ b_c,
                                                 const float* __restrict__ b_o,
                                                 float* __restrict__ out) {
    __shared__ __align__(16) char smem[SMEM_BYTES];

    const int tid  = threadIdx.x;
    const int wave = tid >> 5;
    const int lane = tid & 31;
    const int wm   = wave >> 2;     // 0..1  (M)
    const int wn   = wave & 3;      // 0..3  (N)
    const int m0   = blockIdx.x * BM;
    const int n0   = blockIdx.y * BN;

    const int lrow  = lane & 15;    // index inside 16x16 fragment
    const int lhalf = lane >> 4;    // 0/1: K-half for A, M-half for C/D

    v8f acc[4][2] = {};

    // Prologue: stage tile 0 into buffer 0.
    stage_tile(A, Bm, smem, 0, 0, m0, n0, tid);

    for (int it = 0; it < NIT; ++it) {
        const int cur = it & 1;
        if (it + 1 < NIT)
            stage_tile(A, Bm, smem, cur ^ 1, (it + 1) * BK, m0, n0, tid);
#if HAVE_ASYNC_LDS
        if (it + 1 < NIT) WAIT_ASYNC(4); else WAIT_ASYNC(0);
#endif
        __syncthreads();   // tile `it` visible to all waves

        const __bf16* As = (const __bf16*)(smem + AS_OFF(cur));
        const __bf16* Bs = (const __bf16*)(smem + BS_OFF(cur));

        // A fragments: lane l holds M=l%16, K-chunks at (l/16)*8 and +16.
        v16bf afrag[4];
        #pragma unroll
        for (int fm = 0; fm < 4; ++fm) {
            const __bf16* ap = As + (wm * 64 + fm * 16 + lrow) * LDA + lhalf * 8;
            v8bf lo = *(const v8bf*)(ap);
            v8bf hi = *(const v8bf*)(ap + 16);
            afrag[fm] = __builtin_shufflevector(lo, hi,
                0, 1, 2, 3, 4, 5, 6, 7, 8, 9, 10, 11, 12, 13, 14, 15);
        }
        // B fragments: lane l holds N=l%16, K-chunks at (l/16)*8 and +16.
        v16bf bfrag[2];
        #pragma unroll
        for (int fn = 0; fn < 2; ++fn) {
            const __bf16* bp = Bs + (lhalf * 8) * LDB + wn * 32 + fn * 16 + lrow;
            #pragma unroll
            for (int s = 0; s < 8; ++s) {
                bfrag[fn][s]     = bp[s * LDB];
                bfrag[fn][8 + s] = bp[(16 + s) * LDB];
            }
        }

        #pragma unroll
        for (int fm = 0; fm < 4; ++fm)
            #pragma unroll
            for (int fn = 0; fn < 2; ++fn)
                acc[fm][fn] = __builtin_amdgcn_wmma_f32_16x16x32_bf16(
                    false, afrag[fm], false, bfrag[fn],
                    (short)0, acc[fm][fn], false, false);

        __syncthreads();   // all reads of buffer `cur` done before it is re-staged
    }

    // ---- Stage fp32 accumulator tile in LDS (overlaps the dead A/B buffers).
    float* Ct = (float*)smem;   // [128][128]
    #pragma unroll
    for (int fm = 0; fm < 4; ++fm) {
        #pragma unroll
        for (int fn = 0; fn < 2; ++fn) {
            const int col = wn * 32 + fn * 16 + lrow;
            const int rowbase = wm * 64 + fm * 16 + lhalf * 8;
            #pragma unroll
            for (int v = 0; v < 8; ++v)
                Ct[(rowbase + v) * BN + col] = acc[fm][fn][v];
        }
    }
    __syncthreads();

    // ---- LSTM epilogue: each thread owns one hidden column (j) x 16 rows.
    const int jl = tid & 31;            // local j: 0..31
    const int r0 = tid >> 5;            // 0..7
    const int jg = (n0 >> 2) + jl;      // global hidden column
    const float bfv = b_f[jg];
    const float biv = b_i[jg];
    const float bcv = b_c[jg];
    const float bov = b_o[jg];

    #pragma unroll
    for (int rr = 0; rr < 16; ++rr) {
        const int row = r0 + rr * 8;                 // 0..127
        v4f q = *(const v4f*)(Ct + row * BN + jl * 4);   // (f,i,c,o) gate quad
        const float fg = sigmoid_(q.x + bfv);
        const float ig = sigmoid_(q.y + biv);
        const float cg = tanh_(q.z + bcv);
        const float og = sigmoid_(q.w + bov);
        const size_t gi = (size_t)(m0 + row) * DH + jg;
        const float c2 = fg * c_old[gi] + ig * cg;
        const float hh = og * tanh_(c2);
        out[gi]          = hh;   // h_new (first tuple element)
        out[HN + gi]     = hh;   // h_new (second tuple element)
        out[2 * HN + gi] = c2;   // c_new
    }
}

// ---------------------------------------------------------------------------
// Launch
// ---------------------------------------------------------------------------
extern "C" void kernel_launch(void* const* d_in, const int* in_sizes, int n_in,
                              void* d_out, int out_size, void* d_ws, size_t ws_size,
                              hipStream_t stream) {
    const float* x  = (const float*)d_in[0];
    const float* h  = (const float*)d_in[1];
    const float* c  = (const float*)d_in[2];
    const float* Wf = (const float*)d_in[3];
    const float* bf = (const float*)d_in[4];
    const float* Wi = (const float*)d_in[5];
    const float* bi = (const float*)d_in[6];
    const float* Wc = (const float*)d_in[7];
    const float* bc = (const float*)d_in[8];
    const float* Wo = (const float*)d_in[9];
    const float* bo = (const float*)d_in[10];
    float* out = (float*)d_out;

    char* ws = (char*)d_ws;
    __bf16* xh_bf = (__bf16*)ws;                                   // 16 MB
    __bf16* W_bf  = (__bf16*)(ws + (size_t)16 * 1024 * 1024);      // 16 MB

    pack_xh<<<(B_DIM * K_DIM / 4) / 256, 256, 0, stream>>>(x, h, xh_bf);
    pack_w <<<(K_DIM * N_DIM / 4) / 256, 256, 0, stream>>>(Wf, Wi, Wc, Wo, W_bf);

    dim3 grid(B_DIM / BM, N_DIM / BN);
    gemm_lstm<<<grid, 256, 0, stream>>>(xh_bf, W_bf, c, bf, bi, bc, bo, out);
}
